// EncoderLayer_52192442581853
// MI455X (gfx1250) — matrix operand order repaired and checked by reading, not compiled, not run on hardware
//
#include <hip/hip_runtime.h>

typedef unsigned short u16;
typedef __attribute__((ext_vector_type(16))) __bf16 v16bf;
typedef __attribute__((ext_vector_type(8)))  float  v8f;
typedef __attribute__((ext_vector_type(4)))  unsigned v4u;
typedef __attribute__((ext_vector_type(8)))  unsigned v8u;

#define TOKENS 16384
#define HID    1024
#define NHEAD  8
#define SEQ    4096

__device__ inline u16 f2bf(float f) {
  unsigned u = __float_as_uint(f);
  unsigned r = ((u >> 16) & 1u) + 0x7fffu;   // round-to-nearest-even
  return (u16)((u + r) >> 16);
}

// ---------------------------------------------------------------------------
// f32 -> bf16 conversion
// ---------------------------------------------------------------------------
__global__ void convert_f32_bf16(const float* __restrict__ in,
                                 u16* __restrict__ out, long n) {
  long i = (long)blockIdx.x * blockDim.x + threadIdx.x;
  if (i < n) out[i] = f2bf(in[i]);
}

// ---------------------------------------------------------------------------
// Tiled bf16 WMMA GEMM: C[M,N] = A[M,K](bf16) * B[K,N](bf16) + bias (+epilogue)
// 256 threads = 8 waves (wave32). Block tile 128x128, K-step 32, LDS double-
// buffered.
//   A tile : one TDM tensor_load_to_lds per K-step (wave 0, TENSORcnt)
//   B tile : global_load_b128 + transposing ds_store_b16 scatter
//   bias   : global_load_async_to_lds_b128 once per block (ASYNCcnt)
// Wave grid 2x4; each wave owns a 64x32 tile = 4x2 WMMA accumulators.
// EPI: 0 = bias, 1 = bias then zero where mask[row], 2 = bias + residual.
// Batched over blockIdx.z via element-offset strides (per-head GEMMs).
// ---------------------------------------------------------------------------
template <int EPI>
__global__ __launch_bounds__(256)
void gemm_bf16(const u16* __restrict__ A, const u16* __restrict__ B,
               const float* __restrict__ bias, float* __restrict__ C,
               int M, int N, int K, int lda, int ldb, int ldc,
               long aBatch, long bBatch, long cBatch, long biasBatch,
               const unsigned char* __restrict__ mask,
               const float* __restrict__ residual) {
  __shared__ u16 sA[2][128][32];   // [buf][m][k]
  __shared__ u16 sB[2][128][32];   // [buf][n][k] (transposed)
  __shared__ float sBias[128];

  const int tid  = threadIdx.x;
  const int wave = tid >> 5;
  const int lane = tid & 31;
  const int wm   = wave >> 2;      // 0..1
  const int wn   = wave & 3;       // 0..3
  const int hl   = lane >> 4;      // half-wave select
  const int l16  = lane & 15;

  const int hb = blockIdx.z;
  const u16*   Ab    = A    + (long)hb * aBatch;
  const u16*   Bb    = B    + (long)hb * bBatch;
  float*       Cb    = C    + (long)hb * cBatch;
  const float* biasb = bias + (long)hb * biasBatch;

  const int rowBlk = blockIdx.y * 128;
  const int colBlk = blockIdx.x * 128;

  v8f acc[4][2] = {};

  // B staging assignment (256 threads)
  const int bkrow = tid >> 3;          // 0..31
  const int bncol = (tid & 7) * 16;    // 0..112

  const int nk = K >> 5;

  // --- TDM: DMA the whole 128x32 bf16 A tile into LDS (2D D#, groups 0+1) ---
  auto stageA_tdm = [&](int kt, int buf) {
    unsigned long long ga =
        (unsigned long long)(Ab + (long)rowBlk * lda + (kt << 5));
    unsigned lds = (unsigned)(unsigned long long)&sA[buf][0][0];
    v4u g0;
    g0[0] = 1u;                                   // count=1, user mode
    g0[1] = lds;                                  // lds_addr
    g0[2] = (unsigned)(ga & 0xffffffffu);         // global_addr[31:0]
    g0[3] = (unsigned)((ga >> 32) & 0x01ffffffu)  // global_addr[56:32]
            | (2u << 30);                         // type = 2 (image)
    v8u g1;
    const unsigned td0 = 32, td1 = 128, tile0 = 32, tile1 = 128;
    const unsigned long long st0 = (unsigned long long)lda;
    g1[0] = (1u << 16);                           // data_size = 2 bytes
    g1[1] = (td0 & 0xffffu) << 16;                // tensor_dim0 lo16
    g1[2] = ((td0 >> 16) & 0xffffu) | ((td1 & 0xffffu) << 16);
    g1[3] = ((td1 >> 16) & 0xffffu) | (tile0 << 16);   // tile_dim0
    g1[4] = (tile1 & 0xffffu);                    // tile_dim1; tile_dim2 = 0
    g1[5] = (unsigned)(st0 & 0xffffffffu);        // tensor_dim0_stride lo32
    g1[6] = (unsigned)((st0 >> 32) & 0xffffu);    // stride hi16; dim1_stride=0
    g1[7] = 0u;
    asm volatile("tensor_load_to_lds %0, %1" :: "s"(g0), "s"(g1) : "memory");
  };

  // B tile 32x128 loaded b128 and scattered transposed into sB[n][k]
  auto stageB = [&](int kt, int buf) {
    const u16* pb = Bb + (long)((kt << 5) + bkrow) * ldb + colBlk + bncol;
    uint4 b0 = ((const uint4*)pb)[0];
    uint4 b1 = ((const uint4*)pb)[1];
    union { uint4 v[2]; u16 e[16]; } t;
    t.v[0] = b0; t.v[1] = b1;
#pragma unroll
    for (int j = 0; j < 16; ++j) sB[buf][bncol + j][bkrow] = t.e[j];
    if (kt + 1 < nk)
      __builtin_prefetch(pb + (long)32 * ldb, 0, 0);
  };

  if (wave == 0) {
    // async-copy this block's bias slice (128 f32) into LDS: 1 b128 per lane
    const float* pb = biasb + colBlk + lane * 4;
    unsigned lds = (unsigned)(unsigned long long)&sBias[lane * 4];
    asm volatile("global_load_async_to_lds_b128 %0, %1, off"
                 :: "v"(lds), "v"(pb) : "memory");
    stageA_tdm(0, 0);
  }
  stageB(0, 0);
  asm volatile("s_wait_asynccnt 0x0" ::: "memory");

  for (int kt = 0; kt < nk; ++kt) {
    const int cur = kt & 1;
    if (wave == 0) __builtin_amdgcn_s_wait_tensorcnt(0);
    __syncthreads();   // publish tile kt; orders reads of buf cur^1 vs writes

    if (kt + 1 < nk) {   // overlap next tile's copy with this tile's math
      if (wave == 0) stageA_tdm(kt + 1, cur ^ 1);
      stageB(kt + 1, cur ^ 1);
    }

    // B fragments: 32x16, lane l16 = column; V0-3: K=hl*8.., V4-7: K=16+hl*8..
    v16bf bfrag[2];
#pragma unroll
    for (int tn = 0; tn < 2; ++tn) {
      const u16* pb = &sB[cur][wn * 32 + tn * 16 + l16][0];
      uint4* u = (uint4*)&bfrag[tn];
      u[0] = *(const uint4*)(pb + hl * 8);
      u[1] = *(const uint4*)(pb + 16 + hl * 8);
    }
    // A fragments: 16x32, lane l16 = row; V0-3: K=hl*8.., V4-7: K=16+hl*8..
    v16bf afrag[4];
#pragma unroll
    for (int tm = 0; tm < 4; ++tm) {
      const u16* pa = &sA[cur][wm * 64 + tm * 16 + l16][0];
      uint4* u = (uint4*)&afrag[tm];
      u[0] = *(const uint4*)(pa + hl * 8);
      u[1] = *(const uint4*)(pa + 16 + hl * 8);
    }
#pragma unroll
    for (int tm = 0; tm < 4; ++tm)
#pragma unroll
      for (int tn = 0; tn < 2; ++tn)
        acc[tm][tn] = __builtin_amdgcn_wmma_f32_16x16x32_bf16(
            false, afrag[tm], false, bfrag[tn], (short)0, acc[tm][tn],
            false, false);
  }

  // epilogue: C/D layout -> row = v + 8*hl, col = l16 within each 16x16 tile
#pragma unroll
  for (int tm = 0; tm < 4; ++tm) {
#pragma unroll
    for (int tn = 0; tn < 2; ++tn) {
      const int colL = wn * 32 + tn * 16 + l16;
      const int colG = colBlk + colL;
#pragma unroll
      for (int v = 0; v < 8; ++v) {
        const int rowG = rowBlk + wm * 64 + tm * 16 + v + hl * 8;
        float val = acc[tm][tn][v] + sBias[colL];
        if (EPI == 1) { if (mask[rowG]) val = 0.f; }
        if (EPI == 2) { val += residual[(long)rowG * ldc + colG]; }
        Cb[(long)rowG * ldc + colG] = val;
      }
    }
  }
}

// ---------------------------------------------------------------------------
// Forward / reverse cumulative sums over the sequence axis (in place).
// out_csum layout: [b, s, 4096]; lc = cols [0,2048), rc = cols [2048,4096).
// ---------------------------------------------------------------------------
__global__ void cumsum_kernel(float* __restrict__ cs) {
  int idx = blockIdx.x * blockDim.x + threadIdx.x;  // 0..8191: (b, ch)
  int dir = blockIdx.y;                             // 0 = lc fwd, 1 = rc bwd
  int b  = idx >> 11;
  int ch = idx & 2047;
  long col  = dir ? (2048 + ch) : ch;
  long base = (long)b * SEQ * 4096 + col;
  float acc = 0.f;
  if (dir == 0) {
    for (int s = 0; s < SEQ; ++s) { long o = base + (long)s * 4096; acc += cs[o]; cs[o] = acc; }
  } else {
    for (int s = SEQ - 1; s >= 0; --s) { long o = base + (long)s * 4096; acc += cs[o]; cs[o] = acc; }
  }
}

// ---------------------------------------------------------------------------
// LayerNorm over 2048 (2,NH,IG) for lc_cs / rc_cs.
// c==0 -> gate path -> bf16 into cell_i [t, h, side?256:0 .. +128)
// c==1 -> relu -> f32 lci / rci
// ---------------------------------------------------------------------------
__global__ __launch_bounds__(256)
void ln_lr_kernel(const float* __restrict__ cs,
                  const float* __restrict__ g_l, const float* __restrict__ be_l,
                  const float* __restrict__ g_r, const float* __restrict__ be_r,
                  u16* __restrict__ celli,
                  float* __restrict__ lci, float* __restrict__ rci) {
  const int t    = blockIdx.x;
  const int side = blockIdx.y;   // 0 = lc, 1 = rc
  const int tid  = threadIdx.x;
  const float* G  = side ? g_r  : g_l;
  const float* Be = side ? be_r : be_l;
  const long rowBase = (long)t * 4096 + (side ? 2048 : 0);

  __shared__ float ssum[256], ssq[256];
  float s = 0.f, q = 0.f;
#pragma unroll
  for (int j = 0; j < 8; ++j) {
    float v = cs[rowBase + tid + j * 256];
    s += v; q += v * v;
  }
  ssum[tid] = s; ssq[tid] = q; __syncthreads();
  for (int o = 128; o > 0; o >>= 1) {
    if (tid < o) { ssum[tid] += ssum[tid + o]; ssq[tid] += ssq[tid + o]; }
    __syncthreads();
  }
  const float mean = ssum[0] * (1.f / 2048.f);
  const float var  = ssq[0] * (1.f / 2048.f) - mean * mean;
  const float rstd = rsqrtf(var + 1e-6f);

#pragma unroll
  for (int j = 0; j < 8; ++j) {
    int ch = tid + j * 256;
    float v = cs[rowBase + ch];
    float n = (v - mean) * rstd * G[ch] + Be[ch];
    int c = ch >> 10; int rest = ch & 1023;
    int hh = rest >> 7; int i = rest & 127;
    if (c == 0) {
      celli[(long)t * 3072 + hh * 384 + (side ? 256 : 0) + i] = f2bf(n);
    } else {
      (side ? rci : lci)[(long)t * 1024 + rest] = fmaxf(n, 0.f);
    }
  }
}

// scatter out_x (bf16) into the middle third of each head's cell_i slot
__global__ void scatter_x_kernel(const float* __restrict__ x,
                                 u16* __restrict__ celli, long n) {
  long idx = (long)blockIdx.x * blockDim.x + threadIdx.x;
  if (idx >= n) return;
  long t = idx >> 10; int rest = (int)(idx & 1023);
  int h = rest >> 7; int i = rest & 127;
  celli[t * 3072 + h * 384 + 128 + i] = f2bf(x[idx]);
}

// LayerNorm over 3072 (NH,3,IG) then sigmoid, in place on g
__global__ __launch_bounds__(256)
void ln_sigmoid_kernel(float* __restrict__ g, const float* __restrict__ gg,
                       const float* __restrict__ bg) {
  const int t = blockIdx.x; const int tid = threadIdx.x;
  const long base = (long)t * 3072;
  __shared__ float ssum[256], ssq[256];
  float s = 0.f, q = 0.f;
#pragma unroll
  for (int j = 0; j < 12; ++j) { float v = g[base + tid + j * 256]; s += v; q += v * v; }
  ssum[tid] = s; ssq[tid] = q; __syncthreads();
  for (int o = 128; o > 0; o >>= 1) {
    if (tid < o) { ssum[tid] += ssum[tid + o]; ssq[tid] += ssq[tid + o]; }
    __syncthreads();
  }
  const float mean = ssum[0] * (1.f / 3072.f);
  const float rstd = rsqrtf(ssq[0] * (1.f / 3072.f) - mean * mean + 1e-6f);
#pragma unroll
  for (int j = 0; j < 12; ++j) {
    int ch = tid + j * 256;
    float n = (g[base + ch] - mean) * rstd * gg[ch] + bg[ch];
    g[base + ch] = 1.f / (1.f + __expf(-n));
  }
}

// LayerNorm over 4096 (NH,FG) then relu, output bf16
__global__ __launch_bounds__(256)
void ln_relu_kernel(const float* __restrict__ h, const float* __restrict__ gf,
                    const float* __restrict__ bf, u16* __restrict__ hb) {
  const int t = blockIdx.x; const int tid = threadIdx.x;
  const long base = (long)t * 4096;
  __shared__ float ssum[256], ssq[256];
  float s = 0.f, q = 0.f;
#pragma unroll
  for (int j = 0; j < 16; ++j) { float v = h[base + tid + j * 256]; s += v; q += v * v; }
  ssum[tid] = s; ssq[tid] = q; __syncthreads();
  for (int o = 128; o > 0; o >>= 1) {
    if (tid < o) { ssum[tid] += ssum[tid + o]; ssq[tid] += ssq[tid + o]; }
    __syncthreads();
  }
  const float mean = ssum[0] * (1.f / 4096.f);
  const float rstd = rsqrtf(ssq[0] * (1.f / 4096.f) - mean * mean + 1e-6f);
#pragma unroll
  for (int j = 0; j < 16; ++j) {
    int ch = tid + j * 256;
    float n = (h[base + ch] - mean) * rstd * gf[ch] + bf[ch];
    hb[base + ch] = f2bf(fmaxf(n, 0.f));
  }
}

// out = lci*lcg_g + ig_g*x + rci*rcg_g  -> bf16 for the final GEMM
__global__ void combine_kernel(const float* __restrict__ lci,
                               const float* __restrict__ rci,
                               const float* __restrict__ g,
                               const float* __restrict__ xo,
                               u16* __restrict__ comb, long n) {
  long idx = (long)blockIdx.x * blockDim.x + threadIdx.x;
  if (idx >= n) return;
  long t = idx >> 10; int rest = (int)(idx & 1023);
  int h = rest >> 7; int i = rest & 127;
  long gb = t * 3072 + (long)h * 384 + i;
  float lg = g[gb], ig = g[gb + 128], rg = g[gb + 256];
  float v = lci[idx] * lg + ig * xo[idx] + rci[idx] * rg;
  comb[idx] = f2bf(v);
}

// ---------------------------------------------------------------------------
// Launch
// ---------------------------------------------------------------------------
extern "C" void kernel_launch(void* const* d_in, const int* in_sizes, int n_in,
                              void* d_out, int out_size, void* d_ws, size_t ws_size,
                              hipStream_t stream) {
  (void)in_sizes; (void)n_in; (void)out_size; (void)ws_size;

  const float* inputs = (const float*)d_in[0];
  const unsigned char* mask = (const unsigned char*)d_in[1];
  const float* W_csum = (const float*)d_in[2];
  const float* b_csum = (const float*)d_in[3];
  const float* W_x    = (const float*)d_in[4];
  const float* b_x    = (const float*)d_in[5];
  const float* g_l    = (const float*)d_in[6];
  const float* be_l   = (const float*)d_in[7];
  const float* g_r    = (const float*)d_in[8];
  const float* be_r   = (const float*)d_in[9];
  const float* g_g    = (const float*)d_in[10];
  const float* be_g   = (const float*)d_in[11];
  const float* g_f    = (const float*)d_in[12];
  const float* be_f   = (const float*)d_in[13];
  const float* W_g    = (const float*)d_in[14];
  const float* b_g    = (const float*)d_in[15];
  const float* W_f1   = (const float*)d_in[16];
  const float* b_f1   = (const float*)d_in[17];
  const float* W_f2   = (const float*)d_in[18];
  const float* b_f2   = (const float*)d_in[19];
  const float* W_o    = (const float*)d_in[20];
  const float* b_o    = (const float*)d_in[21];
  float* out = (float*)d_out;

  char* ws = (char*)d_ws;
  size_t off = 0;
  auto alloc = [&](size_t bytes) -> void* {
    void* p = ws + off;
    off = (off + bytes + 255) & ~(size_t)255;
    return p;
  };

  u16*   Xbf     = (u16*)alloc((size_t)TOKENS * HID * 2);
  u16*   Wcsum_b = (u16*)alloc((size_t)HID * 4096 * 2);
  u16*   Wx_b    = (u16*)alloc((size_t)HID * HID * 2);
  u16*   Wo_b    = (u16*)alloc((size_t)HID * HID * 2);
  u16*   Wg_b    = (u16*)alloc((size_t)NHEAD * 384 * 384 * 2);
  u16*   Wf1_b   = (u16*)alloc((size_t)NHEAD * 384 * 512 * 2);
  u16*   Wf2_b   = (u16*)alloc((size_t)NHEAD * 512 * 128 * 2);
  float* csum    = (float*)alloc((size_t)TOKENS * 4096 * 4);  // reused as h
  float* xf      = (float*)alloc((size_t)TOKENS * HID * 4);
  float* lci     = (float*)alloc((size_t)TOKENS * HID * 4);
  float* rci     = (float*)alloc((size_t)TOKENS * HID * 4);
  u16*   celli   = (u16*)alloc((size_t)TOKENS * 3072 * 2);
  float* gbuf    = (float*)alloc((size_t)TOKENS * 3072 * 4);
  u16*   hbf     = (u16*)alloc((size_t)TOKENS * 4096 * 2);
  float* xo      = (float*)alloc((size_t)TOKENS * HID * 4);
  u16*   comb    = (u16*)alloc((size_t)TOKENS * HID * 2);

  dim3 blk(256);
  auto cvt = [&](const float* src, u16* dst, long n) {
    convert_f32_bf16<<<dim3((unsigned)((n + 255) / 256)), blk, 0, stream>>>(src, dst, n);
  };
  cvt(inputs, Xbf, (long)TOKENS * HID);
  cvt(W_csum, Wcsum_b, (long)HID * 4096);
  cvt(W_x,    Wx_b,    (long)HID * HID);
  cvt(W_o,    Wo_b,    (long)HID * HID);
  cvt(W_g,    Wg_b,    (long)NHEAD * 384 * 384);
  cvt(W_f1,   Wf1_b,   (long)NHEAD * 384 * 512);
  cvt(W_f2,   Wf2_b,   (long)NHEAD * 512 * 128);

  // out_csum = X @ W_csum + b, masked rows zeroed
  gemm_bf16<1><<<dim3(4096 / 128, TOKENS / 128, 1), blk, 0, stream>>>(
      Xbf, Wcsum_b, b_csum, csum, TOKENS, 4096, HID, HID, 4096, 4096,
      0, 0, 0, 0, mask, nullptr);
  // out_x = X @ W_x + b
  gemm_bf16<0><<<dim3(HID / 128, TOKENS / 128, 1), blk, 0, stream>>>(
      Xbf, Wx_b, b_x, xf, TOKENS, HID, HID, HID, HID, HID,
      0, 0, 0, 0, nullptr, nullptr);

  cumsum_kernel<<<dim3(32, 2), blk, 0, stream>>>(csum);
  ln_lr_kernel<<<dim3(TOKENS, 2), blk, 0, stream>>>(csum, g_l, be_l, g_r, be_r,
                                                    celli, lci, rci);
  scatter_x_kernel<<<dim3((TOKENS * HID) / 256), blk, 0, stream>>>(
      xf, celli, (long)TOKENS * HID);

  // g = cell_i @ W_g[h] + b_g[h]   (batched over heads)
  gemm_bf16<0><<<dim3(384 / 128, TOKENS / 128, NHEAD), blk, 0, stream>>>(
      celli, Wg_b, b_g, gbuf, TOKENS, 384, 384, 3072, 384, 3072,
      384, (long)384 * 384, 384, 384, nullptr, nullptr);
  ln_sigmoid_kernel<<<dim3(TOKENS), blk, 0, stream>>>(gbuf, g_g, be_g);

  // h = cell_i @ W_f1[h] + b_f1[h]  -> reuse csum buffer
  gemm_bf16<0><<<dim3(512 / 128, TOKENS / 128, NHEAD), blk, 0, stream>>>(
      celli, Wf1_b, b_f1, csum, TOKENS, 512, 384, 3072, 512, 4096,
      384, (long)384 * 512, 512, 512, nullptr, nullptr);
  ln_relu_kernel<<<dim3(TOKENS), blk, 0, stream>>>(csum, g_f, be_f, hbf);

  // x = h @ W_f2[h] + b_f2[h]
  gemm_bf16<0><<<dim3(128 / 128, TOKENS / 128, NHEAD), blk, 0, stream>>>(
      hbf, Wf2_b, b_f2, xo, TOKENS, 128, 512, 4096, 128, 1024,
      512, (long)512 * 128, 128, 128, nullptr, nullptr);

  combine_kernel<<<dim3((TOKENS * HID) / 256), blk, 0, stream>>>(
      lci, rci, gbuf, xo, comb, (long)TOKENS * HID);

  // out = comb @ W_o + b_o + inputs
  gemm_bf16<2><<<dim3(HID / 128, TOKENS / 128, 1), blk, 0, stream>>>(
      comb, Wo_b, b_o, out, TOKENS, HID, HID, HID, HID, HID,
      0, 0, 0, 0, nullptr, inputs);
}
